// EMACodebook_67242007986787
// MI455X (gfx1250) — matrix operand order repaired and checked by reading, not compiled, not run on hardware
//
#include <hip/hip_runtime.h>
#include <math.h>

typedef __bf16 bf16_t;
typedef __attribute__((ext_vector_type(16))) __bf16 v16bf;
typedef __attribute__((ext_vector_type(8)))  float  v8f;
typedef __attribute__((ext_vector_type(8)))  unsigned int v8u;

#define DECAYF 0.99f
#define ONE_MINUS_DECAYF 0.01f
#define EPSF 1e-5f
#define BETAF 0.25f
#define DD 256   // embedding dim (reference D)

static __device__ __forceinline__ bf16_t f2bf(float f) { return (bf16_t)f; }

// ---------------------------------------------------------------------------
// 1) weight prep: bf16 copy + row norms  (one block per codebook row)
// ---------------------------------------------------------------------------
__global__ __launch_bounds__(256) void wprep_kernel(const float* __restrict__ w,
                                                    bf16_t* __restrict__ wb,
                                                    float* __restrict__ wn) {
    const int k = blockIdx.x;
    const size_t base = (size_t)k * DD;
    float v = w[base + threadIdx.x];
    wb[base + threadIdx.x] = f2bf(v);
    float s = v * v;
    #pragma unroll
    for (int m = 16; m >= 1; m >>= 1) s += __shfl_xor(s, m, 32);
    __shared__ float red[8];
    if ((threadIdx.x & 31) == 0) red[threadIdx.x >> 5] = s;
    __syncthreads();
    if (threadIdx.x == 0) {
        float t = 0.f;
        #pragma unroll
        for (int i = 0; i < 8; ++i) t += red[i];
        wn[k] = t;
    }
}

// ---------------------------------------------------------------------------
// 2) zero scratch region
// ---------------------------------------------------------------------------
__global__ __launch_bounds__(256) void zero_kernel(float* __restrict__ p, size_t n) {
    size_t stride = (size_t)gridDim.x * blockDim.x;
    for (size_t i = (size_t)blockIdx.x * blockDim.x + threadIdx.x; i < n; i += stride)
        p[i] = 0.f;
}

// ---------------------------------------------------------------------------
// 3) WMMA bf16 distance + argmin.
//    Block = 256 threads = 8 waves; wave w owns rows [blk*128 + w*16, +16).
//    A-fragments (z rows, 16x256 bf16) resident in VGPRs across the K sweep.
//    Argmin: per-lane running (min, col) in the hot loop (branchless selects
//    only), single cross-lane shuffle reduction after the sweep.
// ---------------------------------------------------------------------------
__global__ __launch_bounds__(256) void argmin_kernel(const float* __restrict__ z,
                                                     const bf16_t* __restrict__ wb,
                                                     const float* __restrict__ wn,
                                                     int* __restrict__ idx,
                                                     int K) {
    const int lane = threadIdx.x & 31;
    const int wave = threadIdx.x >> 5;
    const int half = (lane >= 16) ? 1 : 0;
    const int lrow = lane & 15;
    const int r0   = blockIdx.x * 128 + wave * 16;
    const int zrow = r0 + lrow;

    // --- load A fragments: 8 chunks covering D=256, ISA 16-bit A layout ---
    // lane<16 : row=lrow, K = {c*32+0..7} (VGPR0-3 pairs), {c*32+16..23} (VGPR4-7)
    // lane>=16: row=lrow, K = {c*32+8..15},                {c*32+24..31}
    v16bf a[8];
    const float* zr = z + (size_t)zrow * DD;
    #pragma unroll
    for (int c = 0; c < 8; ++c) {
        const int kb = c * 32 + (half ? 8 : 0);
        #pragma unroll
        for (int r = 0; r < 4; ++r) {
            a[c][2 * r]     = f2bf(zr[kb + 2 * r]);
            a[c][2 * r + 1] = f2bf(zr[kb + 2 * r + 1]);
            a[c][8 + 2 * r]     = f2bf(zr[kb + 16 + 2 * r]);
            a[c][8 + 2 * r + 1] = f2bf(zr[kb + 16 + 2 * r + 1]);
        }
    }

    float bestv[8];
    int   bestc[8];
    #pragma unroll
    for (int v = 0; v < 8; ++v) { bestv[v] = 3.4e38f; bestc[v] = 0; }

    for (int c0 = 0; c0 < K; c0 += 16) {
        const int col = c0 + lrow;                 // this lane's code column
        const bf16_t* wr = wb + (size_t)col * DD;
        const float wnv = wn[col];

        v8f acc = {};
        #pragma unroll
        for (int c = 0; c < 8; ++c) {
            // B layout (32x16 bf16): lanes 0-15 hold K 0-7,16-23 of the chunk,
            // lanes 16-31 hold K 8-15,24-31; VGPR r = pair (2r, 2r+1).
            const int kb = c * 32 + (half ? 8 : 0);
            const uint4 blo = *(const uint4*)(wr + kb);       // 16B
            const uint4 bhi = *(const uint4*)(wr + kb + 16);  // 16B
            v8u braw = {blo.x, blo.y, blo.z, blo.w, bhi.x, bhi.y, bhi.z, bhi.w};
            v16bf b = __builtin_bit_cast(v16bf, braw);
            acc = __builtin_amdgcn_wmma_f32_16x16x32_bf16(
                false, a[c], false, b, (short)0, acc, false, false);
        }

        // C layout: VGPR v holds (row=v, col=lane) for lanes 0-15 and
        // (row=v+8, col=lane-16) for lanes 16-31. Per-lane running min only;
        // columns visited by a lane are ascending, so strict < keeps earliest.
        #pragma unroll
        for (int v = 0; v < 8; ++v) {
            float d = __builtin_fmaf(-2.0f, acc[v], wnv);
            bool  t = d < bestv[v];
            bestv[v] = t ? d   : bestv[v];
            bestc[v] = t ? col : bestc[v];
        }
    }

    // cross-lane min+argmin within each 16-lane half (once, after the sweep)
    #pragma unroll
    for (int v = 0; v < 8; ++v) {
        float dist = bestv[v];
        int   cc   = bestc[v];
        #pragma unroll
        for (int m = 1; m < 16; m <<= 1) {
            float od = __shfl_xor(dist, m, 32);
            int   oc = __shfl_xor(cc,   m, 32);
            bool  t  = (od < dist) || (od == dist && oc < cc);
            dist = t ? od : dist;
            cc   = t ? oc : cc;
        }
        bestc[v] = cc;
    }

    if (lrow == 0) {   // lane 0 -> rows r0+0..7, lane 16 -> rows r0+8..15
        #pragma unroll
        for (int v = 0; v < 8; ++v) idx[r0 + half * 8 + v] = bestc[v];
    }
}

// ---------------------------------------------------------------------------
// 4) cluster counts
// ---------------------------------------------------------------------------
__global__ __launch_bounds__(256) void counts_kernel(const int* __restrict__ idx,
                                                     int* __restrict__ counts, int N) {
    int n = blockIdx.x * blockDim.x + threadIdx.x;
    if (n < N) atomicAdd(&counts[idx[n]], 1);
}

// ---------------------------------------------------------------------------
// 5) dw = one_hot(idx)^T @ z : scatter-add z rows into dw[idx[n], :]
// ---------------------------------------------------------------------------
__global__ __launch_bounds__(256) void dw_kernel(const float* __restrict__ z,
                                                 const int* __restrict__ idx,
                                                 float* __restrict__ dw, size_t total) {
    size_t stride = (size_t)gridDim.x * blockDim.x;
    for (size_t i = (size_t)blockIdx.x * blockDim.x + threadIdx.x; i < total; i += stride) {
        size_t n = i >> 8;
        int    d = (int)(i & (DD - 1));
        atomicAdd(&dw[(size_t)idx[n] * DD + d], z[i]);
    }
}

// ---------------------------------------------------------------------------
// 6) EMA cluster sizes (+Laplace normalize) and perplexity. Single block.
// ---------------------------------------------------------------------------
__global__ __launch_bounds__(1024) void cs_kernel(const float* __restrict__ ecs,
                                                  const int* __restrict__ counts,
                                                  float* __restrict__ out_cs,
                                                  float* __restrict__ out_plex,
                                                  int K, float invN) {
    __shared__ float rn[32], rp[32], tot[2];
    float nsum = 0.f, psum = 0.f;
    for (int k = threadIdx.x; k < K; k += blockDim.x) {
        float cnt = (float)counts[k];
        float cs  = ecs[k] * DECAYF + ONE_MINUS_DECAYF * cnt;
        out_cs[k] = cs;                 // temporarily raw
        nsum += cs;
        float p = cnt * invN;
        psum += p * logf(p + 1e-10f);
    }
    #pragma unroll
    for (int m = 16; m >= 1; m >>= 1) {
        nsum += __shfl_xor(nsum, m, 32);
        psum += __shfl_xor(psum, m, 32);
    }
    int wid = threadIdx.x >> 5, ln = threadIdx.x & 31;
    if (ln == 0) { rn[wid] = nsum; rp[wid] = psum; }
    __syncthreads();
    if (threadIdx.x == 0) {
        float tn = 0.f, tp = 0.f;
        int nw = blockDim.x >> 5;
        for (int i = 0; i < nw; ++i) { tn += rn[i]; tp += rp[i]; }
        tot[0] = tn; tot[1] = tp;
    }
    __syncthreads();
    float n = tot[0];
    float scale = n / (n + (float)K * EPSF);
    for (int k = threadIdx.x; k < K; k += blockDim.x)
        out_cs[k] = (out_cs[k] + EPSF) * scale;   // same thread wrote these
    if (threadIdx.x == 0) *out_plex = expf(-tot[1]);
}

// ---------------------------------------------------------------------------
// 7) new_ema_w and new_weight
// ---------------------------------------------------------------------------
__global__ __launch_bounds__(256) void ema_kernel(const float* __restrict__ ema_w,
                                                  const float* __restrict__ dw,
                                                  const float* __restrict__ cs,
                                                  float* __restrict__ o_emaw,
                                                  float* __restrict__ o_w, size_t total) {
    size_t stride = (size_t)gridDim.x * blockDim.x;
    for (size_t i = (size_t)blockIdx.x * blockDim.x + threadIdx.x; i < total; i += stride) {
        int k = (int)(i >> 8);
        float e = ema_w[i] * DECAYF + ONE_MINUS_DECAYF * dw[i];
        o_emaw[i] = e;
        o_w[i] = e / cs[k];
    }
}

// ---------------------------------------------------------------------------
// 8) z_q gather (straight-through == z_q numerically), idx-as-float, loss sum
// ---------------------------------------------------------------------------
__global__ __launch_bounds__(256) void zq_kernel(const float* __restrict__ z,
                                                 const float* __restrict__ wq,
                                                 const int* __restrict__ idx,
                                                 float* __restrict__ o_zq,
                                                 float* __restrict__ o_idx,
                                                 float* __restrict__ lossacc,
                                                 size_t total) {
    float lsum = 0.f;
    size_t stride = (size_t)gridDim.x * blockDim.x;
    for (size_t i = (size_t)blockIdx.x * blockDim.x + threadIdx.x; i < total; i += stride) {
        size_t n = i >> 8;
        int    d = (int)(i & (DD - 1));
        int    k = idx[n];
        float  q = wq[(size_t)k * DD + d];
        o_zq[i] = q;
        if (d == 0) o_idx[n] = (float)k;
        float df = q - z[i];
        lsum += df * df;
    }
    #pragma unroll
    for (int m = 16; m >= 1; m >>= 1) lsum += __shfl_xor(lsum, m, 32);
    __shared__ float red[8];
    int wid = threadIdx.x >> 5, ln = threadIdx.x & 31;
    if (ln == 0) red[wid] = lsum;
    __syncthreads();
    if (threadIdx.x == 0) {
        float t = 0.f;
        #pragma unroll
        for (int i = 0; i < 8; ++i) t += red[i];
        atomicAdd(lossacc, t);
    }
}

// ---------------------------------------------------------------------------
// 9) finalize loss:  (1 + beta) * mean((z_q - z)^2)
// ---------------------------------------------------------------------------
__global__ void fin_kernel(const float* __restrict__ lossacc,
                           float* __restrict__ o_loss, float invTot) {
    *o_loss = (1.0f + BETAF) * (*lossacc) * invTot;
}

// ---------------------------------------------------------------------------
extern "C" void kernel_launch(void* const* d_in, const int* in_sizes, int n_in,
                              void* d_out, int out_size, void* d_ws, size_t ws_size,
                              hipStream_t stream) {
    const float* z      = (const float*)d_in[0];   // [N, 256]
    const float* weight = (const float*)d_in[1];   // [K, 256]
    const float* ema_w  = (const float*)d_in[2];   // [K, 256]
    const float* ema_cs = (const float*)d_in[3];   // [K]

    const int N = in_sizes[0] / DD;
    const int K = in_sizes[3];
    const size_t ND = (size_t)N * DD;
    const size_t KD = (size_t)K * DD;

    // ---- output layout (tuple concatenated, all as float) ----
    float* o      = (float*)d_out;
    float* o_zq   = o;              // N*D
    float* o_idx  = o_zq + ND;      // N
    float* o_loss = o_idx + N;      // 1
    float* o_plex = o_loss + 1;     // 1
    float* o_cs   = o_plex + 1;     // K
    float* o_emaw = o_cs + K;       // K*D
    float* o_w    = o_emaw + KD;    // K*D

    // ---- workspace layout ----
    char* w8 = (char*)d_ws;
    bf16_t* wb     = (bf16_t*)w8;                   size_t off = KD * 2;
    float*  wn     = (float*)(w8 + off);            off += (size_t)K * 4;
    int*    counts = (int*)(w8 + off);              off += (size_t)K * 4;
    float*  dwbuf  = (float*)(w8 + off);            off += KD * 4;
    int*    idxb   = (int*)(w8 + off);              off += (size_t)N * 4;
    float*  lossac = (float*)(w8 + off);

    // zero counts + dw + idx + lossacc region (as floats)
    size_t zero_n = (size_t)K + KD + (size_t)N + 1;

    wprep_kernel<<<K, 256, 0, stream>>>(weight, wb, wn);
    zero_kernel<<<2048, 256, 0, stream>>>((float*)counts, zero_n);
    argmin_kernel<<<N / 128, 256, 0, stream>>>(z, wb, wn, idxb, K);
    counts_kernel<<<(N + 255) / 256, 256, 0, stream>>>(idxb, counts, N);
    dw_kernel<<<4096, 256, 0, stream>>>(z, idxb, dwbuf, ND);
    cs_kernel<<<1, 1024, 0, stream>>>(ema_cs, counts, o_cs, o_plex, K, 1.0f / (float)N);
    ema_kernel<<<2048, 256, 0, stream>>>(ema_w, dwbuf, o_cs, o_emaw, o_w, KD);
    zq_kernel<<<4096, 256, 0, stream>>>(z, o_w, idxb, o_zq, o_idx, lossac, ND);
    fin_kernel<<<1, 1, 0, stream>>>(lossac, o_loss, 1.0f / (float)ND);
}
